// Qwen3MoeSparseMoeBlock_31645319037665
// MI455X (gfx1250) — matrix operand order repaired and checked by reading, not compiled, not run on hardware
//
#include <hip/hip_runtime.h>
#include <hip/hip_bf16.h>
#include <math.h>

// ---------------------------------------------------------------------------
// Qwen3 MoE sparse block for MI455X (gfx1250, wave32, WMMA).
// bf16 inputs -> v_wmma_f32_16x16x32_bf16, f32 accumulate (compute-bound:
// 103 GFLOP vs ~4.3us of HBM traffic at 23.3 TB/s; weights L2-resident).
// Down-projection fused into ONE dense GEMM by folding the per-(token,expert)
// combine weight into Hmid': out = Hmid'[T, E*I] @ w_down[E*I, H].
// GEMM staging uses GLOBAL_LOAD_ASYNC_TO_LDS_B128 (ASYNCcnt) when available.
// Workspace layout (bytes):
//   X16   @ 0        : T*H bf16          = 4 MB
//   WG16  @ 4  MB    : E*H*I bf16        = 16 MB
//   WU16  @ 20 MB    : E*H*I bf16        = 16 MB
//   WD16  @ 36 MB    : E*I*H bf16        = 16 MB
//   HM16  @ 52 MB    : T*E*I bf16        = 32 MB
//   COMB  @ 84 MB    : T*E f32           = 128 KB
// ---------------------------------------------------------------------------

typedef __attribute__((ext_vector_type(16))) __bf16 bf16x16;
typedef __attribute__((ext_vector_type(8)))  __bf16 bf16x8;
typedef __attribute__((ext_vector_type(4)))  __bf16 bf16x4;
typedef __attribute__((ext_vector_type(8)))  float  f32x8;
typedef __attribute__((ext_vector_type(4)))  float  f32x4;

constexpr int Tt = 2048;    // B*S tokens
constexpr int Hd = 1024;    // hidden
constexpr int Id = 512;     // moe intermediate
constexpr int Ne = 16;      // experts
constexpr int KI = Ne * Id; // 8192 stacked K for down-proj

// ------------------------- async global -> LDS ------------------------------

#if __has_builtin(__builtin_amdgcn_global_load_async_to_lds_b128)
#define USE_ASYNC_LDS 1
#else
#define USE_ASYNC_LDS 0
#endif

__device__ __forceinline__ void wait_async0() {
#if __has_builtin(__builtin_amdgcn_s_wait_asynccnt)
  __builtin_amdgcn_s_wait_asynccnt(0);
#else
  asm volatile("s_wait_asynccnt 0" ::: "memory");
#endif
}

#if USE_ASYNC_LDS
// The builtin's pointee type is a 16-byte int vector (per clang diagnostic:
// "__attribute__((__vector_size__(4 * sizeof(int)))) int __device__ *").
typedef int v4i_gcc __attribute__((vector_size(16)));

// 16-byte async copy: global (AS1) -> LDS (AS3). The low 32 bits of a flat
// LDS pointer are the LDS offset (shared aperture lives in addr[63:32]), so
// the truncating inttoptr yields the addrspace(3) value the HW needs.
__device__ __forceinline__ void async_cp16(const __bf16* g, __bf16* l) {
  __builtin_amdgcn_global_load_async_to_lds_b128(
      (__attribute__((address_space(1))) v4i_gcc*)(uintptr_t)g,
      (__attribute__((address_space(3))) v4i_gcc*)(uint32_t)(uintptr_t)l,
      0, 0);
}
#endif

// ----------------------------- WMMA helpers --------------------------------

__device__ __forceinline__ f32x8 wmma_bf16(bf16x16 a, bf16x16 b, f32x8 c) {
  // (neg_a, A, neg_b, B, c_mod, C, reuse_a, reuse_b)
  return __builtin_amdgcn_wmma_f32_16x16x32_bf16(false, a, false, b,
                                                 (short)0, c, false, false);
}

// A fragment (16x32 bf16): lane L holds row M = L%16; K-halves depend on L/16:
// lanes 0-15: K 0..7 (VGPR0-3) and 16..23 (VGPR4-7); lanes 16-31: K 8..15, 24..31.
__device__ __forceinline__ bf16x16 load_frag_a(const __bf16* As, int row, int lane) {
  const int klo = (lane < 16) ? 0 : 8;
  union { bf16x16 v; bf16x8 h[2]; } f;
  const __bf16* p = As + row * 32 + klo;
  f.h[0] = *(const bf16x8*)(p);
  f.h[1] = *(const bf16x8*)(p + 16);
  return f.v;
}

// B fragment (32x16 bf16, K x N): lane L holds column N = L%16; K as A fragment:
// lanes 0-15: K 0..7, 16..23; lanes 16-31: K 8..15, 24..31. Bs is row-major [32][64] in LDS.
__device__ __forceinline__ bf16x16 load_frag_b(const __bf16* Bs, int n0, int lane) {
  const int klo = (lane < 16) ? 0 : 8;
  bf16x16 v;
  const __bf16* p = Bs + n0 + (lane & 15);
#pragma unroll
  for (int s = 0; s < 8; ++s) {
    v[s]     = p[(klo + s) * 64];
    v[8 + s] = p[(16 + klo + s) * 64];
  }
  return v;
}

// ----------------------------- fp32 -> bf16 --------------------------------

__global__ __launch_bounds__(256)
void moe_cvt_f32_bf16_kernel(const float* __restrict__ src,
                             __bf16* __restrict__ dst, int n4) {
  int i = blockIdx.x * blockDim.x + threadIdx.x;
  if (i < n4) {
    f32x4 v = ((const f32x4*)src)[i];
    ((bf16x4*)dst)[i] = __builtin_convertvector(v, bf16x4);
  }
}

// ----------------------------- router --------------------------------------

__global__ __launch_bounds__(256)
void moe_router_logits_kernel(const float* __restrict__ x,     // [Tt][Hd]
                              const float* __restrict__ gw,    // [Hd][Ne]
                              float* __restrict__ logits) {    // [Tt][Ne]
  const int lane = threadIdx.x & 31;
  const int wave = threadIdx.x >> 5;
  const int t = blockIdx.x * 8 + wave;
  if (t >= Tt) return;

  float acc[Ne];
#pragma unroll
  for (int e = 0; e < Ne; ++e) acc[e] = 0.0f;

  const float* xr = x + (size_t)t * Hd;
  for (int h = lane; h < Hd; h += 32) {
    float xv = xr[h];
    const float* g = gw + (size_t)h * Ne;
#pragma unroll
    for (int e = 0; e < Ne; ++e) acc[e] = fmaf(xv, g[e], acc[e]);
  }
#pragma unroll
  for (int off = 16; off > 0; off >>= 1) {
#pragma unroll
    for (int e = 0; e < Ne; ++e) acc[e] += __shfl_xor(acc[e], off, 32);
  }
  if (lane < Ne) logits[(size_t)t * Ne + lane] = acc[lane];
}

__global__ __launch_bounds__(256)
void moe_router_topk_kernel(const float* __restrict__ logits,
                            float* __restrict__ comb) {        // [Tt][Ne]
  int t = blockIdx.x * blockDim.x + threadIdx.x;
  if (t >= Tt) return;
  float p[Ne];
  float mx = -3.0e38f;
#pragma unroll
  for (int e = 0; e < Ne; ++e) { p[e] = logits[(size_t)t * Ne + e]; mx = fmaxf(mx, p[e]); }
  float s = 0.0f;
#pragma unroll
  for (int e = 0; e < Ne; ++e) { p[e] = __expf(p[e] - mx); s += p[e]; }
  float inv = 1.0f / s;
#pragma unroll
  for (int e = 0; e < Ne; ++e) p[e] *= inv;

  unsigned sel = 0; float wsum = 0.0f;
  for (int k = 0; k < 8; ++k) {
    int best = 0; float bv = -1.0f;
#pragma unroll
    for (int e = 0; e < Ne; ++e)
      if (!((sel >> e) & 1u) && p[e] > bv) { bv = p[e]; best = e; }
    sel |= 1u << best; wsum += bv;
  }
  float invw = 1.0f / wsum;
#pragma unroll
  for (int e = 0; e < Ne; ++e)
    comb[(size_t)t * Ne + e] = ((sel >> e) & 1u) ? p[e] * invw : 0.0f;
}

// ----------------------------- GEMM 1: gate/up + SiLU ----------------------
// Per expert e (gridDim.z): Hmid'[t, e*I + n] = silu(x@wg)*(x@wu)*ce, bf16.
// Tile: 128(M) x 64(N), K-step 32, double-buffered LDS, 8 waves of 32x32.

__global__ __launch_bounds__(256)
void moe_gemm1_kernel(const __bf16* __restrict__ X,    // [Tt][Hd]
                      const __bf16* __restrict__ WG,   // [Ne][Hd][Id]
                      const __bf16* __restrict__ WU,   // [Ne][Hd][Id]
                      const float*  __restrict__ comb, // [Tt][Ne]
                      __bf16* __restrict__ HM) {       // [Tt][KI]
  const int e  = blockIdx.z;
  const int m0 = blockIdx.y * 128;
  const int n0 = blockIdx.x * 64;
  const int tid  = threadIdx.x;
  const int lane = tid & 31;
  const int wid  = tid >> 5;
  const int wm = (wid & 3) * 32;   // wave M offset in tile
  const int wn = (wid >> 2) * 32;  // wave N offset in tile

  __shared__ __align__(16) __bf16 As[2][128 * 32];
  __shared__ __align__(16) __bf16 Bg[2][32 * 64];
  __shared__ __align__(16) __bf16 Bu[2][32 * 64];

  const __bf16* WGe = WG + (size_t)e * Hd * Id;
  const __bf16* WUe = WU + (size_t)e * Hd * Id;

  // staging: A = 512 16B chunks (2/thread); Bg,Bu = 256 chunks (1/thread each)
  const int ar0 = tid >> 2,          ac0 = (tid & 3) * 8;
  const int ar1 = (tid + 256) >> 2,  ac1 = (tid & 3) * 8;
  const int br  = tid >> 3,          bc  = (tid & 7) * 8;

  f32x8 accG[2][2] = {}; f32x8 accU[2][2] = {};
  const int KT = Hd / 32;

#if USE_ASYNC_LDS
  auto stage = [&](int buf, int k) {
    async_cp16(X   + (size_t)(m0 + ar0) * Hd + k + ac0, &As[buf][ar0 * 32 + ac0]);
    async_cp16(X   + (size_t)(m0 + ar1) * Hd + k + ac1, &As[buf][ar1 * 32 + ac1]);
    async_cp16(WGe + (size_t)(k + br) * Id + n0 + bc,   &Bg[buf][br * 64 + bc]);
    async_cp16(WUe + (size_t)(k + br) * Id + n0 + bc,   &Bu[buf][br * 64 + bc]);
  };
  stage(0, 0);
  wait_async0();
  __syncthreads();
#else
  bf16x8 ra0, ra1, rg, ru;
  auto gload = [&](int k) {
    ra0 = *(const bf16x8*)(X + (size_t)(m0 + ar0) * Hd + k + ac0);
    ra1 = *(const bf16x8*)(X + (size_t)(m0 + ar1) * Hd + k + ac1);
    rg  = *(const bf16x8*)(WGe + (size_t)(k + br) * Id + n0 + bc);
    ru  = *(const bf16x8*)(WUe + (size_t)(k + br) * Id + n0 + bc);
  };
  auto sstore = [&](int buf) {
    *(bf16x8*)(&As[buf][ar0 * 32 + ac0]) = ra0;
    *(bf16x8*)(&As[buf][ar1 * 32 + ac1]) = ra1;
    *(bf16x8*)(&Bg[buf][br * 64 + bc])   = rg;
    *(bf16x8*)(&Bu[buf][br * 64 + bc])   = ru;
  };
  gload(0);
  sstore(0);
  __syncthreads();
#endif

  for (int kt = 0; kt < KT; ++kt) {
    const int cur = kt & 1;
#if USE_ASYNC_LDS
    if (kt + 1 < KT) stage(cur ^ 1, (kt + 1) * 32);
#else
    if (kt + 1 < KT) gload((kt + 1) * 32);
#endif
    if (kt + 2 < KT) {  // global_prefetch_b8 two tiles ahead
      __builtin_prefetch(X + (size_t)(m0 + ar0) * Hd + (kt + 2) * 32 + ac0, 0, 3);
      __builtin_prefetch(WGe + (size_t)((kt + 2) * 32 + br) * Id + n0 + bc, 0, 3);
      __builtin_prefetch(WUe + (size_t)((kt + 2) * 32 + br) * Id + n0 + bc, 0, 3);
    }
    bf16x16 a0 = load_frag_a(&As[cur][0], wm + (lane & 15), lane);
    bf16x16 a1 = load_frag_a(&As[cur][0], wm + 16 + (lane & 15), lane);
    bf16x16 g0 = load_frag_b(&Bg[cur][0], wn,      lane);
    bf16x16 g1 = load_frag_b(&Bg[cur][0], wn + 16, lane);
    bf16x16 u0 = load_frag_b(&Bu[cur][0], wn,      lane);
    bf16x16 u1 = load_frag_b(&Bu[cur][0], wn + 16, lane);
    accG[0][0] = wmma_bf16(a0, g0, accG[0][0]);
    accG[0][1] = wmma_bf16(a0, g1, accG[0][1]);
    accG[1][0] = wmma_bf16(a1, g0, accG[1][0]);
    accG[1][1] = wmma_bf16(a1, g1, accG[1][1]);
    accU[0][0] = wmma_bf16(a0, u0, accU[0][0]);
    accU[0][1] = wmma_bf16(a0, u1, accU[0][1]);
    accU[1][0] = wmma_bf16(a1, u0, accU[1][0]);
    accU[1][1] = wmma_bf16(a1, u1, accU[1][1]);
#if USE_ASYNC_LDS
    wait_async0();
#else
    if (kt + 1 < KT) sstore(cur ^ 1);
#endif
    __syncthreads();
  }

  // epilogue: hmid' = silu(g)*u*ce -> bf16. C layout: VGPR r, lane L:
  // M = r + 8*(L>=16), N = L%16.
  const int rlo = (lane >= 16) ? 8 : 0;
#pragma unroll
  for (int i = 0; i < 2; ++i) {
#pragma unroll
    for (int r = 0; r < 8; ++r) {
      const int row = m0 + wm + 16 * i + r + rlo;
      const float ce = comb[(size_t)row * Ne + e];
#pragma unroll
      for (int j = 0; j < 2; ++j) {
        const int col = n0 + wn + 16 * j + (lane & 15);
        const float g = accG[i][j][r];
        const float u = accU[i][j][r];
        const float val = (g / (1.0f + __expf(-g))) * u * ce;
        HM[(size_t)row * KI + (size_t)e * Id + col] = (__bf16)val;
      }
    }
  }
}

// ----------------------------- GEMM 2: down-proj ----------------------------
// out[2048,1024] = HM[2048,8192] @ WD[8192,1024], f32 out, overwrite.

__global__ __launch_bounds__(256)
void moe_gemm2_kernel(const __bf16* __restrict__ HM,  // [Tt][KI]
                      const __bf16* __restrict__ WD,  // [KI][Hd]
                      float* __restrict__ out) {      // [Tt][Hd]
  const int m0 = blockIdx.y * 128;
  const int n0 = blockIdx.x * 64;
  const int tid  = threadIdx.x;
  const int lane = tid & 31;
  const int wid  = tid >> 5;
  const int wm = (wid & 3) * 32;
  const int wn = (wid >> 2) * 32;

  __shared__ __align__(16) __bf16 As[2][128 * 32];
  __shared__ __align__(16) __bf16 Bs[2][32 * 64];

  const int ar0 = tid >> 2,          ac0 = (tid & 3) * 8;
  const int ar1 = (tid + 256) >> 2,  ac1 = (tid & 3) * 8;
  const int br  = tid >> 3,          bc  = (tid & 7) * 8;

  f32x8 acc[2][2] = {};
  const int KT = KI / 32;  // 256

#if USE_ASYNC_LDS
  auto stage = [&](int buf, int k) {
    async_cp16(HM + (size_t)(m0 + ar0) * KI + k + ac0, &As[buf][ar0 * 32 + ac0]);
    async_cp16(HM + (size_t)(m0 + ar1) * KI + k + ac1, &As[buf][ar1 * 32 + ac1]);
    async_cp16(WD + (size_t)(k + br) * Hd + n0 + bc,   &Bs[buf][br * 64 + bc]);
  };
  stage(0, 0);
  wait_async0();
  __syncthreads();
#else
  bf16x8 ra0, ra1, rb;
  auto gload = [&](int k) {
    ra0 = *(const bf16x8*)(HM + (size_t)(m0 + ar0) * KI + k + ac0);
    ra1 = *(const bf16x8*)(HM + (size_t)(m0 + ar1) * KI + k + ac1);
    rb  = *(const bf16x8*)(WD + (size_t)(k + br) * Hd + n0 + bc);
  };
  auto sstore = [&](int buf) {
    *(bf16x8*)(&As[buf][ar0 * 32 + ac0]) = ra0;
    *(bf16x8*)(&As[buf][ar1 * 32 + ac1]) = ra1;
    *(bf16x8*)(&Bs[buf][br * 64 + bc])   = rb;
  };
  gload(0);
  sstore(0);
  __syncthreads();
#endif

  for (int kt = 0; kt < KT; ++kt) {
    const int cur = kt & 1;
#if USE_ASYNC_LDS
    if (kt + 1 < KT) stage(cur ^ 1, (kt + 1) * 32);
#else
    if (kt + 1 < KT) gload((kt + 1) * 32);
#endif
    if (kt + 2 < KT) {
      __builtin_prefetch(HM + (size_t)(m0 + ar0) * KI + (kt + 2) * 32 + ac0, 0, 3);
      __builtin_prefetch(WD + (size_t)((kt + 2) * 32 + br) * Hd + n0 + bc, 0, 3);
    }
    bf16x16 a0 = load_frag_a(&As[cur][0], wm + (lane & 15), lane);
    bf16x16 a1 = load_frag_a(&As[cur][0], wm + 16 + (lane & 15), lane);
    bf16x16 b0 = load_frag_b(&Bs[cur][0], wn,      lane);
    bf16x16 b1 = load_frag_b(&Bs[cur][0], wn + 16, lane);
    acc[0][0] = wmma_bf16(a0, b0, acc[0][0]);
    acc[0][1] = wmma_bf16(a0, b1, acc[0][1]);
    acc[1][0] = wmma_bf16(a1, b0, acc[1][0]);
    acc[1][1] = wmma_bf16(a1, b1, acc[1][1]);
#if USE_ASYNC_LDS
    wait_async0();
#else
    if (kt + 1 < KT) sstore(cur ^ 1);
#endif
    __syncthreads();
  }

  const int rlo = (lane >= 16) ? 8 : 0;
#pragma unroll
  for (int i = 0; i < 2; ++i) {
#pragma unroll
    for (int r = 0; r < 8; ++r) {
      const int row = m0 + wm + 16 * i + r + rlo;
#pragma unroll
      for (int j = 0; j < 2; ++j) {
        const int col = n0 + wn + 16 * j + (lane & 15);
        out[(size_t)row * Hd + col] = acc[i][j][r];
      }
    }
  }
}

// ----------------------------- host launcher --------------------------------

extern "C" void kernel_launch(void* const* d_in, const int* in_sizes, int n_in,
                              void* d_out, int out_size, void* d_ws, size_t ws_size,
                              hipStream_t stream) {
  const float* x     = (const float*)d_in[0];  // [B,S,H]
  const float* gw    = (const float*)d_in[1];  // [H,E]
  const float* wgate = (const float*)d_in[2];  // [E,H,I]
  const float* wup   = (const float*)d_in[3];  // [E,H,I]
  const float* wdown = (const float*)d_in[4];  // [E,I,H] == stacked [E*I,H]

  float* out    = (float*)d_out;               // [Tt][Hd]
  float* logits = out + (size_t)Tt * Hd;       // [Tt][Ne] (tuple output #2)

  char* ws = (char*)d_ws;
  const size_t MB = 1024u * 1024u;
  __bf16* X16  = (__bf16*)(ws + 0 * MB);
  __bf16* WG16 = (__bf16*)(ws + 4 * MB);
  __bf16* WU16 = (__bf16*)(ws + 20 * MB);
  __bf16* WD16 = (__bf16*)(ws + 36 * MB);
  __bf16* HM16 = (__bf16*)(ws + 52 * MB);
  float*  comb = (float*)(ws + 84 * MB);

  const int thr = 256;
  const int nX = Tt * Hd;          // 2,097,152
  const int nW = Ne * Hd * Id;     // 8,388,608

  moe_cvt_f32_bf16_kernel<<<(nX / 4 + thr - 1) / thr, thr, 0, stream>>>(x, X16, nX / 4);
  moe_cvt_f32_bf16_kernel<<<(nW / 4 + thr - 1) / thr, thr, 0, stream>>>(wgate, WG16, nW / 4);
  moe_cvt_f32_bf16_kernel<<<(nW / 4 + thr - 1) / thr, thr, 0, stream>>>(wup,   WU16, nW / 4);
  moe_cvt_f32_bf16_kernel<<<(nW / 4 + thr - 1) / thr, thr, 0, stream>>>(wdown, WD16, nW / 4);

  moe_router_logits_kernel<<<Tt / 8, 256, 0, stream>>>(x, gw, logits);
  moe_router_topk_kernel<<<Tt / 256, 256, 0, stream>>>(logits, comb);

  moe_gemm1_kernel<<<dim3(Id / 64, Tt / 128, Ne), 256, 0, stream>>>(
      X16, WG16, WU16, comb, HM16);
  moe_gemm2_kernel<<<dim3(Hd / 64, Tt / 128), 256, 0, stream>>>(
      HM16, WD16, out);
}
